// Pose_RWKV_29308856828034
// MI455X (gfx1250) — compile-verified, matrix operand order checked by reading, not run on hardware
//
#include <hip/hip_runtime.h>
#include <cstdint>
#include <cstddef>

// ---------------------------------------------------------------------------
// Pose_RWKV on MI455X (gfx1250):
//   - all GEMMs via v_wmma_f32_16x16x32_bf16 (f32 accumulate)
//   - A/B tiles staged into LDS by the Tensor Data Mover (tensor_load_to_lds),
//     double-buffered so the DMA of tile i+1 overlaps WMMA of tile i
//   - TDM LDS padding (80B rows) kills bank conflicts, keeps 16B alignment
// B=32 T=512 Dv=512 Di=256 F=768 H=3072 BT=16384.
// ---------------------------------------------------------------------------

typedef __attribute__((ext_vector_type(16))) __bf16 v16bf;
typedef __attribute__((ext_vector_type(8)))  float  v8f;
typedef unsigned int v4u __attribute__((ext_vector_type(4)));
typedef int          v8i __attribute__((ext_vector_type(8)));
typedef int          v4i __attribute__((ext_vector_type(4)));

#define BT_  16384
#define B_   32
#define T_   512
#define DV_  512
#define DI_  256
#define F_   768
#define F2_  1536
#define H_   3072

// ------------------------------ convert f32 -> bf16 ------------------------
__global__ void cvt_f32_bf16(const float* __restrict__ src,
                             __bf16* __restrict__ dst, size_t n) {
  size_t i = (size_t)blockIdx.x * blockDim.x + threadIdx.x;
  if (i < n) dst[i] = (__bf16)src[i];
}

// ---------------- transpose-convert: W[K][N] f32 -> Wt[N][K] bf16 ----------
__global__ void cvt_t_f32_bf16(const float* __restrict__ src,
                               __bf16* __restrict__ dst, int K, int N) {
  size_t i = (size_t)blockIdx.x * blockDim.x + threadIdx.x;
  if (i >= (size_t)K * N) return;
  int n = (int)(i / K), k = (int)(i % K);
  dst[i] = (__bf16)src[(size_t)k * N + n];
}

// ------------------------------ pack feat = [fv | fi] as bf16 --------------
__global__ void pack_feat(const float* __restrict__ fv,
                          const float* __restrict__ fi,
                          __bf16* __restrict__ dst) {
  size_t i = (size_t)blockIdx.x * blockDim.x + threadIdx.x;
  if (i >= (size_t)BT_ * F_) return;
  int m = (int)(i / F_), c = (int)(i % F_);
  float f = (c < DV_) ? fv[(size_t)m * DV_ + c] : fi[(size_t)m * DI_ + (c - DV_)];
  dst[i] = (__bf16)f;
}

// ------------------------------ gumbel hard gate ---------------------------
// forward of straight-through: x = feat * (argmax_j(logit+gumbel) == 0)
__global__ void gate_kernel(const float* __restrict__ fv,
                            const float* __restrict__ fi,
                            const float* __restrict__ logits,  // [BT, 2F]
                            const float* __restrict__ gum,     // [BT, 2F]
                            float* __restrict__ X) {
  size_t i = (size_t)blockIdx.x * blockDim.x + threadIdx.x;
  if (i >= (size_t)BT_ * F_) return;
  int m = (int)(i / F_), c = (int)(i % F_);
  float f = (c < DV_) ? fv[(size_t)m * DV_ + c] : fi[(size_t)m * DI_ + (c - DV_)];
  size_t li = (size_t)m * F2_ + 2 * c;
  float a0 = logits[li] + gum[li];
  float a1 = logits[li + 1] + gum[li + 1];
  X[i] = (a0 >= a1) ? f : 0.0f;
}

// ------------------------------ LN + token-shift + mix ---------------------
__device__ __forceinline__ float block_reduce(float v, float* sh) {
  int tid = threadIdx.x;
  sh[tid] = v; __syncthreads();
  for (int s = 128; s > 0; s >>= 1) {
    if (tid < s) sh[tid] += sh[tid + s];
    __syncthreads();
  }
  float r = sh[0]; __syncthreads();
  return r;
}

__global__ __launch_bounds__(256) void ln_shift_mix(
    const float* __restrict__ X,
    const float* __restrict__ g, const float* __restrict__ b,
    const float* __restrict__ mk, const float* __restrict__ mv,
    const float* __restrict__ mr,
    __bf16* __restrict__ ok, __bf16* __restrict__ ov, __bf16* __restrict__ orr) {
  __shared__ float sh[256];
  int m = blockIdx.x;
  int t = m % T_;
  int tid = threadIdx.x;
  const float* row = X + (size_t)m * F_;
  const float* prow = row - F_;

  float xc[3], xp[3];
#pragma unroll
  for (int j = 0; j < 3; ++j) {
    int c = tid + j * 256;
    xc[j] = row[c];
    xp[j] = (t > 0) ? prow[c] : 0.0f;
  }
  float s   = block_reduce(xc[0] + xc[1] + xc[2], sh);
  float ss  = block_reduce(xc[0]*xc[0] + xc[1]*xc[1] + xc[2]*xc[2], sh);
  float sp  = block_reduce(xp[0] + xp[1] + xp[2], sh);
  float ssp = block_reduce(xp[0]*xp[0] + xp[1]*xp[1] + xp[2]*xp[2], sh);

  const float invC = 1.0f / (float)F_;
  float mean  = s * invC;
  float var   = ss * invC - mean * mean;
  float rstd  = rsqrtf(var + 1e-5f);
  float meanp = sp * invC;
  float varp  = ssp * invC - meanp * meanp;
  float rstdp = rsqrtf(varp + 1e-5f);

#pragma unroll
  for (int j = 0; j < 3; ++j) {
    int c = tid + j * 256;
    size_t o = (size_t)m * F_ + c;
    float xn = (xc[j] - mean) * rstd * g[c] + b[c];
    float xs = (t > 0) ? ((xp[j] - meanp) * rstdp * g[c] + b[c]) : 0.0f;
    float k_ = mk[c];
    ok[o] = (__bf16)(xn * k_ + xs * (1.0f - k_));
    if (ov)  { float v_ = mv[c]; ov[o]  = (__bf16)(xn * v_ + xs * (1.0f - v_)); }
    if (orr) { float r_ = mr[c]; orr[o] = (__bf16)(xn * r_ + xs * (1.0f - r_)); }
  }
}

// ------------------------------ WKV scan (fused with sigmoid(r)) -----------
__global__ void wkv_kernel(const float* __restrict__ K,
                           const float* __restrict__ V,
                           const float* __restrict__ Rp,
                           const float* __restrict__ time_decay,
                           const float* __restrict__ time_first,
                           __bf16* __restrict__ O) {
  int idx = blockIdx.x * blockDim.x + threadIdx.x;
  if (idx >= B_ * F_) return;
  int bb_ = idx / F_, c = idx % F_;
  float w = -expf(time_decay[c]);
  float u = time_first[c];
  float aa = 0.0f, bb = 0.0f, pp = -1e38f;
  for (int t = 0; t < T_; ++t) {
    size_t base = ((size_t)(bb_ * T_ + t)) * F_ + c;
    float kt = K[base], vt = V[base];
    float ww = u + kt;
    float p  = fmaxf(pp, ww);
    float e1 = expf(pp - p), e2 = expf(ww - p);
    float out = (e1 * aa + e2 * vt) / (e1 * bb + e2);
    float r = 1.0f / (1.0f + expf(-Rp[base]));
    O[base] = (__bf16)(r * out);
    float ww2 = pp + w;
    float p2  = fmaxf(ww2, kt);
    float f1 = expf(ww2 - p2), f2 = expf(kt - p2);
    aa = f1 * aa + f2 * vt;
    bb = f1 * bb + f2;
    pp = p2;
  }
}

// ------------------------------ elementwise combine: X += a*b --------------
__global__ void combine_mul_add(float* __restrict__ X,
                                const float* __restrict__ a,
                                const float* __restrict__ bch, size_t n) {
  size_t i = (size_t)blockIdx.x * blockDim.x + threadIdx.x;
  if (i < n) X[i] += a[i] * bch[i];
}

// ------------------------------ TDM 2D tile -> LDS -------------------------
// Loads a [rows x 32] bf16 tile (row stride = `stride` elements) starting at
// `g` into LDS at byte offset lds_off. TDM pad: +4 dwords after every 16
// dwords (64B row) -> LDS row stride 80B (bank-conflict-free, 16B aligned).
// D# layout per CDNA5 ISA ch.8 (group0: count/lds/global/type, group1: dims).
#define LROW 40  // padded LDS row stride in bf16 elements (80 bytes)

__device__ __forceinline__ void tdm_load_tile(uint32_t lds_off,
                                              const __bf16* g,
                                              uint32_t rows,
                                              uint32_t stride_elems,
                                              uint32_t tensor_rows) {
  uint64_t ga = (uint64_t)(uintptr_t)g;
  v4u g0;
  g0[0] = 1u;                                   // count=1, user descriptor
  g0[1] = lds_off;                              // lds_addr
  g0[2] = (uint32_t)ga;                         // global_addr[31:0]
  g0[3] = (uint32_t)((ga >> 32) & 0x01FFFFFFu)  // global_addr[56:32]
          | (2u << 30);                         // type=2 ("image")
  uint32_t td0 = stride_elems;                  // tensor_dim0 (OOB bound)
  uint32_t td1 = tensor_rows;                   // tensor_dim1
  v8i g1;
  g1[0] = (int)((1u << 16)        // data_size = 2 bytes
              | (1u << 20)        // pad_enable
              | (3u << 22)        // pad_interval: 16 dwords (64B)
              | (3u << 25));      // pad_amount: 4 dwords (16B)
  g1[1] = (int)((td0 & 0xFFFFu) << 16);                    // tensor_dim0 lo16
  g1[2] = (int)((td0 >> 16) | ((td1 & 0xFFFFu) << 16));    // td0 hi / td1 lo
  g1[3] = (int)((td1 >> 16) | (32u << 16));                // td1 hi / tile_dim0=32
  g1[4] = (int)rows;                                       // tile_dim1 (tile_dim2=0)
  g1[5] = (int)stride_elems;                               // tensor_dim0_stride lo32
  g1[6] = 0;                                               // stride hi / dim1_stride lo
  g1[7] = 0;
  v4i g2 = {0, 0, 0, 0};
  v4i g3 = {0, 0, 0, 0};
#if defined(__clang_major__) && (__clang_major__ >= 23)
  v8i g4 = {0, 0, 0, 0, 0, 0, 0, 0};
  __builtin_amdgcn_tensor_load_to_lds(g0, g1, g2, g3, g4, 0);
#else
  __builtin_amdgcn_tensor_load_to_lds(g0, g1, g2, g3, 0);
#endif
}

// ------------------------------ WMMA GEMM ----------------------------------
// out[M,N] = act(A[M,K] @ Wt[N,K]^T + bias) (+ Cin). bf16 operands, f32 acc.
// Block 256 threads = 8 waves; tile 64(M) x 128(N); K-step 32.
// Wave (wm 0..3, wn 0..1) -> 16 rows x 64 cols = four 16x16x32 WMMA accums.
// Double-buffered TDM staging: DMA of tile i+1 overlaps WMMA of tile i.
#define GBM 64
#define GBN 128
#define GBK 32

__global__ __launch_bounds__(256) void gemm_bf16_wmma(
    const __bf16* __restrict__ A, const __bf16* __restrict__ Wt,
    const float* __restrict__ bias, const float* __restrict__ Cin,
    float* __restrict__ out, int M, int N, int K, int act) {
  __shared__ __align__(16) __bf16 sA[2][GBM * LROW];   // [row][k] padded
  __shared__ __align__(16) __bf16 sB[2][GBN * LROW];   // [col][k] padded

  const int tid  = threadIdx.x;
  const int m0   = blockIdx.y * GBM;
  const int n0   = blockIdx.x * GBN;
  const int lane = tid & 31;
  const int wv   = __builtin_amdgcn_readfirstlane(tid >> 5);  // uniform wave id
  const int wm   = wv & 3,    wn = wv >> 2;
  const int half = lane >> 4, lr = lane & 15;

  uint32_t ldsA[2] = { (uint32_t)(uintptr_t)&sA[0][0],
                       (uint32_t)(uintptr_t)&sA[1][0] };
  uint32_t ldsB[2] = { (uint32_t)(uintptr_t)&sB[0][0],
                       (uint32_t)(uintptr_t)&sB[1][0] };

  const __bf16* Abase = A  + (size_t)m0 * K;
  const __bf16* Bbase = Wt + (size_t)n0 * K;

  v8f acc[4];
#pragma unroll
  for (int t2 = 0; t2 < 4; ++t2)
#pragma unroll
    for (int i = 0; i < 8; ++i) acc[t2][i] = 0.0f;

  // prologue: stage tile 0 into buffer 0
  if (wv == 0) tdm_load_tile(ldsA[0], Abase, GBM, (uint32_t)K, (uint32_t)M);
  if (wv == 1) tdm_load_tile(ldsB[0], Bbase, GBN, (uint32_t)K, (uint32_t)N);
  __builtin_amdgcn_s_wait_tensorcnt(0);
  __syncthreads();

  int buf = 0;
  for (int k0 = 0; k0 < K; k0 += GBK) {
    // --- prefetch tile i+1 into the other buffer (free since barrier @ i-1)
    int nk = k0 + GBK;
    if (nk < K) {
      if (wv == 0) tdm_load_tile(ldsA[buf ^ 1], Abase + nk, GBM,
                                 (uint32_t)K, (uint32_t)M);
      if (wv == 1) tdm_load_tile(ldsB[buf ^ 1], Bbase + nk, GBN,
                                 (uint32_t)K, (uint32_t)N);
    }

    // --- compute on current buffer while DMA is in flight
    const __bf16* cA = &sA[buf][0];
    const __bf16* cB = &sB[buf][0];
    union Fr { v16bf v; uint32_t u[8]; };
    Fr fa;
#pragma unroll
    for (int d = 0; d < 8; ++d) {
      int kb = ((d < 4) ? (2 * d) : (16 + 2 * (d - 4))) + half * 8;
      fa.u[d] = *reinterpret_cast<const uint32_t*>(&cA[(wm * 16 + lr) * LROW + kb]);
    }
#pragma unroll
    for (int t2 = 0; t2 < 4; ++t2) {
      Fr fb;
      int nc = wn * 64 + t2 * 16 + lr;
#pragma unroll
      for (int d = 0; d < 8; ++d) {
        int kb = ((d < 4) ? (2 * d) : (16 + 2 * (d - 4))) + half * 8;
        fb.u[d] = *reinterpret_cast<const uint32_t*>(&cB[nc * LROW + kb]);
      }
      acc[t2] = __builtin_amdgcn_wmma_f32_16x16x32_bf16(
          false, fa.v, false, fb.v, (short)0, acc[t2], false, false);
    }

    // --- next tile complete (issuing waves) + everyone done reading `buf`
    __builtin_amdgcn_s_wait_tensorcnt(0);
    __syncthreads();
    buf ^= 1;
  }

  // epilogue; C/D layout: lane lr = N col, vgpr i -> M row = half*8 + i
#pragma unroll
  for (int t2 = 0; t2 < 4; ++t2) {
    int n = n0 + wn * 64 + t2 * 16 + lr;
    float bvv = bias ? bias[n] : 0.0f;
#pragma unroll
    for (int i = 0; i < 8; ++i) {
      int m = m0 + wm * 16 + half * 8 + i;
      float vv = acc[t2][i] + bvv;
      if (act == 1)      { vv = (vv > 0.0f) ? vv * vv : 0.0f; }       // relu^2
      else if (act == 2) { vv = 1.0f / (1.0f + expf(-vv)); }          // sigmoid
      else if (act == 3) { vv = (vv > 0.0f) ? vv : 0.1f * vv; }       // leaky
      if (Cin) vv += Cin[(size_t)m * N + n];
      out[(size_t)m * N + n] = vv;
    }
  }
}

// ------------------------------ tiny head: [BT,128] @ [128,6] + b ----------
__global__ void reg2_kernel(const float* __restrict__ h,
                            const float* __restrict__ W2,
                            const float* __restrict__ b2,
                            float* __restrict__ out) {
  int tid = blockIdx.x * blockDim.x + threadIdx.x;
  if (tid >= BT_ * 6) return;
  int m = tid / 6, j = tid % 6;
  float s = b2[j];
  const float* hr = h + (size_t)m * 128;
#pragma unroll 8
  for (int k = 0; k < 128; ++k) s += hr[k] * W2[k * 6 + j];
  out[tid] = s;
}

// ---------------------------------------------------------------------------
extern "C" void kernel_launch(void* const* d_in, const int* in_sizes, int n_in,
                              void* d_out, int out_size, void* d_ws, size_t ws_size,
                              hipStream_t stream) {
  (void)in_sizes; (void)n_in; (void)out_size; (void)ws_size;
  const float* fv       = (const float*)d_in[0];
  const float* fi       = (const float*)d_in[1];
  const float* fuse_W   = (const float*)d_in[2];
  const float* fuse_b   = (const float*)d_in[3];
  const float* gumbel   = (const float*)d_in[4];
  const float* ln1_g    = (const float*)d_in[5];
  const float* ln1_b    = (const float*)d_in[6];
  const float* mix_k    = (const float*)d_in[7];
  const float* mix_v    = (const float*)d_in[8];
  const float* mix_r    = (const float*)d_in[9];
  const float* tdecay   = (const float*)d_in[10];
  const float* tfirst   = (const float*)d_in[11];
  const float* Wk       = (const float*)d_in[12];
  const float* Wv       = (const float*)d_in[13];
  const float* Wr       = (const float*)d_in[14];
  const float* Wo       = (const float*)d_in[15];
  const float* ln2_g    = (const float*)d_in[16];
  const float* ln2_b    = (const float*)d_in[17];
  const float* cmix_k   = (const float*)d_in[18];
  const float* cmix_r   = (const float*)d_in[19];
  const float* cWk      = (const float*)d_in[20];
  const float* cWv      = (const float*)d_in[21];
  const float* cWr      = (const float*)d_in[22];
  const float* reg_W1   = (const float*)d_in[23];
  const float* reg_b1   = (const float*)d_in[24];
  const float* reg_W2   = (const float*)d_in[25];
  const float* reg_b2   = (const float*)d_in[26];
  float* out = (float*)d_out;

  // ---------------- workspace layout (bytes) ----------------
  char* ws = (char*)d_ws;
  const size_t SZ_F32 = (size_t)BT_ * F_ * 4;
  const size_t SZ_H32 = (size_t)BT_ * H_ * 4;
  size_t off = 0;
  float*  X   = (float*)(ws + off); off += SZ_F32;
  float*  A0  = (float*)(ws + off); off += SZ_H32;   // logits / kk
  float*  A1  = (float*)(ws + off); off += SZ_F32;   // k / kv
  float*  A2  = (float*)(ws + off); off += SZ_F32;   // v
  float*  A3  = (float*)(ws + off); off += SZ_F32;   // r_pre / sigmoid(r)
  float*  HB  = (float*)(ws + off); off += (size_t)BT_ * 128 * 4;
  __bf16* BFA = (__bf16*)(ws + off); off += (size_t)BT_ * H_ * 2;
  __bf16* BFW = (__bf16*)(ws + off);

  // bf16 TRANSPOSED weight regions (each stored [N][K])
  __bf16* bW_fuse = BFW;                               // [1536][768]
  __bf16* bW_k    = bW_fuse + (size_t)F_ * F2_;        // [768][768]
  __bf16* bW_v    = bW_k    + (size_t)F_ * F_;
  __bf16* bW_r    = bW_v    + (size_t)F_ * F_;
  __bf16* bW_o    = bW_r    + (size_t)F_ * F_;
  __bf16* bW_ck   = bW_o    + (size_t)F_ * F_;         // [3072][768]
  __bf16* bW_cv   = bW_ck   + (size_t)F_ * H_;         // [768][3072]
  __bf16* bW_cr   = bW_cv   + (size_t)H_ * F_;
  __bf16* bW_r1   = bW_cr   + (size_t)F_ * F_;         // [128][768]

  auto cvt = [&](const float* s, __bf16* d, size_t n) {
    cvt_f32_bf16<<<(unsigned)((n + 255) / 256), 256, 0, stream>>>(s, d, n);
  };
  auto cvtT = [&](const float* s, __bf16* d, int K, int N) {
    size_t n = (size_t)K * N;
    cvt_t_f32_bf16<<<(unsigned)((n + 255) / 256), 256, 0, stream>>>(s, d, K, N);
  };
  auto gemm = [&](const __bf16* Aop, const __bf16* Wop, const float* bias,
                  const float* Cin, float* outp, int N, int K, int act) {
    dim3 grid((unsigned)(N / GBN), (unsigned)(BT_ / GBM));
    gemm_bf16_wmma<<<grid, 256, 0, stream>>>(Aop, Wop, bias, Cin, outp,
                                             BT_, N, K, act);
  };

  const size_t nBF = (size_t)BT_ * F_;
  const size_t nBH = (size_t)BT_ * H_;
  const unsigned gBF = (unsigned)((nBF + 255) / 256);

  // ---- weights -> bf16, transposed to [N][K]
  cvtT(fuse_W, bW_fuse, F_, F2_);
  cvtT(Wk,  bW_k,  F_, F_);
  cvtT(Wv,  bW_v,  F_, F_);
  cvtT(Wr,  bW_r,  F_, F_);
  cvtT(Wo,  bW_o,  F_, F_);
  cvtT(cWk, bW_ck, F_, H_);
  cvtT(cWv, bW_cv, H_, F_);
  cvtT(cWr, bW_cr, F_, F_);
  cvtT(reg_W1, bW_r1, F_, 128);

  // ---- fusion + gumbel hard gate
  pack_feat<<<gBF, 256, 0, stream>>>(fv, fi, BFA);
  gemm(BFA, bW_fuse, fuse_b, nullptr, A0, F2_, F_, 0);          // logits
  gate_kernel<<<gBF, 256, 0, stream>>>(fv, fi, A0, gumbel, X);

  // ---- RWKV time-mixing
  __bf16* bfXK = BFA;
  __bf16* bfXV = BFA + nBF;
  __bf16* bfXR = BFA + 2 * nBF;
  ln_shift_mix<<<BT_, 256, 0, stream>>>(X, ln1_g, ln1_b, mix_k, mix_v, mix_r,
                                        bfXK, bfXV, bfXR);
  gemm(bfXK, bW_k, nullptr, nullptr, A1, F_, F_, 0);            // k
  gemm(bfXV, bW_v, nullptr, nullptr, A2, F_, F_, 0);            // v
  gemm(bfXR, bW_r, nullptr, nullptr, A3, F_, F_, 0);            // r pre-sigmoid
  wkv_kernel<<<(B_ * F_ + 255) / 256, 256, 0, stream>>>(A1, A2, A3, tdecay,
                                                        tfirst, BFA);
  gemm(BFA, bW_o, nullptr, X, X, F_, F_, 0);                    // x += (r*wkv)@Wo

  // ---- RWKV channel-mixing
  ln_shift_mix<<<BT_, 256, 0, stream>>>(X, ln2_g, ln2_b, cmix_k, cmix_r, cmix_r,
                                        bfXK, nullptr, bfXR);
  gemm(bfXK, bW_ck, nullptr, nullptr, A0, H_, F_, 1);           // kk = relu^2
  gemm(bfXR, bW_cr, nullptr, nullptr, A3, F_, F_, 2);           // sigmoid(xr@cWr)
  cvt(A0, BFA, nBH);                                            // kk -> bf16
  gemm(BFA, bW_cv, nullptr, nullptr, A1, F_, H_, 0);            // kk@cWv
  combine_mul_add<<<gBF, 256, 0, stream>>>(X, A3, A1, nBF);     // x += sig*kv

  // ---- regressor head
  cvt(X, BFA, nBF);
  gemm(BFA, bW_r1, reg_b1, nullptr, HB, 128, F_, 3);            // leaky(x@W1+b1)
  reg2_kernel<<<(BT_ * 6 + 255) / 256, 256, 0, stream>>>(HB, reg_W2, reg_b2, out);
}
